// Bengio03HighwayBiLm_71021579206699
// MI455X (gfx1250) — matrix operand order repaired and checked by reading, not compiled, not run on hardware
//
#include <hip/hip_runtime.h>
#include <hip/hip_bf16.h>

// ---------------- problem constants ----------------
#define Bn 32
#define Sn 512
#define Dn 512
#define Wn 2
#define SP (Sn + 2 * Wn)         // 516 padded rows
#define INSZ (Dn * (Wn + 1))     // 1536
#define MTOT (Bn * Sn)           // 16384 rows

typedef __attribute__((ext_vector_type(16))) __bf16 v16bf;
typedef __attribute__((ext_vector_type(8)))  float  v8f;

union Frag { v16bf v; uint4 q[2]; };

__device__ __forceinline__ __bf16 f2bf(float f) {
  union { float f; unsigned u; } x; x.f = f;
  unsigned short s = (unsigned short)((x.u + 0x7FFFu + ((x.u >> 16) & 1u)) >> 16);
  union { unsigned short s; __bf16 b; } o; o.s = s;
  return o.b;
}
__device__ __forceinline__ float bf2f(__bf16 b) {
  union { __bf16 b; unsigned short s; } i; i.b = b;
  union { unsigned u; float f; } o; o.u = ((unsigned)i.s) << 16;
  return o.f;
}

// ---------------- conversion kernels ----------------
__global__ void cvt_f32_bf16(const float* __restrict__ src,
                             __bf16* __restrict__ dst, int n) {
  int i = blockIdx.x * 256 + threadIdx.x;
  if (i < n) dst[i] = f2bf(src[i]);
}

// Build padded bf16 activation tensor P[B, 516, 512]
__global__ void pad_kernel(const float* __restrict__ inp,
                           const float* __restrict__ lp,
                           const float* __restrict__ rp,
                           __bf16* __restrict__ P) {
  int i = blockIdx.x * 256 + threadIdx.x;
  if (i >= Bn * SP * Dn) return;
  int d = i & (Dn - 1);
  int r = (i >> 9) % SP;
  int b = i / (SP * Dn);
  float v;
  if (r < Wn)            v = lp[r * Dn + d];
  else if (r < Sn + Wn)  v = inp[((size_t)b * Sn + (r - Wn)) * Dn + d];
  else                   v = rp[(r - Sn - Wn) * Dn + d];
  P[i] = f2bf(v);
}

// ---------------- window projection GEMM + ReLU ----------------
// C[m, n] = relu( sum_k win[m,k] * W[n,k] + bias[n] ),  K = 1536
// win row m = contiguous slice of P starting at (b*516 + s + 2*side)*512.
// Block: 256 thr (8 waves). Tile 128M x 128N. Wave: 32M x 64N (2x4 accums).
// Double-buffered LDS, register-staged global loads (pipelined).
__global__ __launch_bounds__(256) void win_gemm(
    const __bf16* __restrict__ P,
    const __bf16* __restrict__ Wl, const __bf16* __restrict__ Wr,
    const float* __restrict__ bl,  const float* __restrict__ br,
    __bf16* __restrict__ X0) {
  __shared__ __align__(16) __bf16 lA[2][128][40];
  __shared__ __align__(16) __bf16 lB[2][128][40];

  const int side = blockIdx.z;
  const int m0 = blockIdx.x * 128;
  const int n0 = blockIdx.y * 128;
  const __bf16* Wt   = side ? Wr : Wl;
  const float*  bias = side ? br : bl;
  const int t = threadIdx.x;
  const int lane = t & 31, wave = t >> 5;
  const int mW = (wave >> 1) * 32;     // wave M offset (4 groups)
  const int nW = (wave & 1) * 64;      // wave N offset (2 groups)
  const int r = lane & 15, h = lane >> 4;

  // staging: each thread owns 2 A-chunks and 2 B-chunks of 8 bf16 (16B each)
  const int row0 = t >> 2;             // 0..63
  const int row1 = row0 + 64;          // 64..127
  const int col8 = (t & 3) * 8;

  const int mA0 = m0 + row0, mA1 = m0 + row1;
  const __bf16* pA0 =
      P + (((size_t)(mA0 >> 9) * SP + (mA0 & (Sn - 1)) + 2 * side) << 9) + col8;
  const __bf16* pA1 =
      P + (((size_t)(mA1 >> 9) * SP + (mA1 & (Sn - 1)) + 2 * side) << 9) + col8;
  const __bf16* pB0 = Wt + (size_t)(n0 + row0) * INSZ + col8;
  const __bf16* pB1 = Wt + (size_t)(n0 + row1) * INSZ + col8;

  v8f acc[2][4];
#pragma unroll
  for (int i = 0; i < 2; i++)
#pragma unroll
    for (int j = 0; j < 4; j++)
#pragma unroll
      for (int v = 0; v < 8; v++) acc[i][j][v] = 0.f;

  // prologue: fetch + stage tile 0
  uint4 ra0 = *(const uint4*)pA0, ra1 = *(const uint4*)pA1;
  uint4 rb0 = *(const uint4*)pB0, rb1 = *(const uint4*)pB1;
  pA0 += 32; pA1 += 32; pB0 += 32; pB1 += 32;
  *(uint4*)&lA[0][row0][col8] = ra0;
  *(uint4*)&lA[0][row1][col8] = ra1;
  *(uint4*)&lB[0][row0][col8] = rb0;
  *(uint4*)&lB[0][row1][col8] = rb1;

  const int NK = INSZ / 32;            // 48 K-chunks
  int buf = 0;
  for (int kk = 0; kk < NK; ++kk) {
    __syncthreads();                   // LDS[buf] visible to all waves
    const bool more = (kk + 1) < NK;
    if (more) {                        // issue next tile's loads early
      ra0 = *(const uint4*)pA0; ra1 = *(const uint4*)pA1;
      rb0 = *(const uint4*)pB0; rb1 = *(const uint4*)pB1;
      pA0 += 32; pA1 += 32; pB0 += 32; pB1 += 32;
    }

    Frag a[2], bfr[4];
#pragma unroll
    for (int mi = 0; mi < 2; mi++) {   // A layout: contiguous K per lane
      a[mi].q[0] = *(const uint4*)&lA[buf][mW + mi * 16 + r][16 * h];
      a[mi].q[1] = *(const uint4*)&lA[buf][mW + mi * 16 + r][16 * h + 8];
    }
#pragma unroll
    for (int ni = 0; ni < 4; ni++) {   // B layout: contiguous K per lane
      bfr[ni].q[0] = *(const uint4*)&lB[buf][nW + ni * 16 + r][16 * h];
      bfr[ni].q[1] = *(const uint4*)&lB[buf][nW + ni * 16 + r][16 * h + 8];
    }
#pragma unroll
    for (int mi = 0; mi < 2; mi++)
#pragma unroll
      for (int ni = 0; ni < 4; ni++)
        acc[mi][ni] = __builtin_amdgcn_wmma_f32_16x16x32_bf16(
            false, a[mi].v, false, bfr[ni].v, (short)0, acc[mi][ni], false, false);

    if (more) {                        // stage next tile behind compute
      *(uint4*)&lA[buf ^ 1][row0][col8] = ra0;
      *(uint4*)&lA[buf ^ 1][row1][col8] = ra1;
      *(uint4*)&lB[buf ^ 1][row0][col8] = rb0;
      *(uint4*)&lB[buf ^ 1][row1][col8] = rb1;
    }
    buf ^= 1;
  }

  __bf16* Xs = X0 + (size_t)side * MTOT * Dn;
#pragma unroll
  for (int mi = 0; mi < 2; mi++)
#pragma unroll
    for (int ni = 0; ni < 4; ni++) {
      int n = n0 + nW + ni * 16 + r;
      float bv = bias[n];
#pragma unroll
      for (int v = 0; v < 8; v++) {
        int m = m0 + mW + mi * 16 + 8 * h + v;
        float y = acc[mi][ni][v] + bv;
        Xs[(size_t)m * Dn + n] = f2bf(y > 0.f ? y : 0.f);
      }
    }
}

// ---------------- highway layer (fused proj + gate) ----------------
// proj = x @ W^T + b  (W: [1024,512]); nl = relu(proj[:, :512]),
// g = sigmoid(proj[:, 512:]); y = g*x + (1-g)*nl
// Block: 256 thr, tile 128M x 64N; each wave: 16M x 64N, dual accums (nl+gate).
// Double-buffered LDS pipeline; final layer streams fp32 out with NT stores.
__global__ __launch_bounds__(256) void highway_layer(
    const __bf16* __restrict__ Xin,
    const __bf16* __restrict__ Wlhw, const __bf16* __restrict__ Wrhw,
    const float* __restrict__ blhw,  const float* __restrict__ brhw,
    __bf16* __restrict__ Xout, float* __restrict__ out, int final_layer) {
  __shared__ __align__(16) __bf16 lA[2][128][40];
  __shared__ __align__(16) __bf16 lB[2][128][40]; // rows 0..63 nl, 64..127 gate

  const int side = blockIdx.z;
  const int m0 = blockIdx.x * 128;
  const int n0 = blockIdx.y * 64;
  const __bf16* Wt   = side ? Wrhw : Wlhw;
  const float*  bias = side ? brhw : blhw;
  const __bf16* Xs = Xin + (size_t)side * MTOT * Dn;
  const int t = threadIdx.x;
  const int lane = t & 31, wave = t >> 5;
  const int r = lane & 15, h = lane >> 4;
  const int mWv = wave * 16;

  const int row0 = t >> 2;             // 0..63  -> B nl rows  W[n0+row0]
  const int row1 = row0 + 64;          // 64..127-> B gate rows W[512+n0+row0]
  const int col8 = (t & 3) * 8;

  const __bf16* pA0 = Xs + (size_t)(m0 + row0) * Dn + col8;
  const __bf16* pA1 = Xs + (size_t)(m0 + row1) * Dn + col8;
  const __bf16* pB0 = Wt + (size_t)(n0 + row0) * Dn + col8;
  const __bf16* pB1 = Wt + (size_t)(Dn + n0 + row0) * Dn + col8;

  v8f accN[4], accG[4];
#pragma unroll
  for (int j = 0; j < 4; j++)
#pragma unroll
    for (int v = 0; v < 8; v++) { accN[j][v] = 0.f; accG[j][v] = 0.f; }

  uint4 ra0 = *(const uint4*)pA0, ra1 = *(const uint4*)pA1;
  uint4 rb0 = *(const uint4*)pB0, rb1 = *(const uint4*)pB1;
  pA0 += 32; pA1 += 32; pB0 += 32; pB1 += 32;
  *(uint4*)&lA[0][row0][col8] = ra0;
  *(uint4*)&lA[0][row1][col8] = ra1;
  *(uint4*)&lB[0][row0][col8] = rb0;
  *(uint4*)&lB[0][row1][col8] = rb1;

  const int NK = Dn / 32;              // 16 K-chunks
  int buf = 0;
  for (int kk = 0; kk < NK; ++kk) {
    __syncthreads();
    const bool more = (kk + 1) < NK;
    if (more) {
      ra0 = *(const uint4*)pA0; ra1 = *(const uint4*)pA1;
      rb0 = *(const uint4*)pB0; rb1 = *(const uint4*)pB1;
      pA0 += 32; pA1 += 32; pB0 += 32; pB1 += 32;
    }

    Frag a, bn[4], bg[4];
    a.q[0] = *(const uint4*)&lA[buf][mWv + r][16 * h];
    a.q[1] = *(const uint4*)&lA[buf][mWv + r][16 * h + 8];
#pragma unroll
    for (int ni = 0; ni < 4; ni++) {
      bn[ni].q[0] = *(const uint4*)&lB[buf][ni * 16 + r][16 * h];
      bn[ni].q[1] = *(const uint4*)&lB[buf][ni * 16 + r][16 * h + 8];
      bg[ni].q[0] = *(const uint4*)&lB[buf][64 + ni * 16 + r][16 * h];
      bg[ni].q[1] = *(const uint4*)&lB[buf][64 + ni * 16 + r][16 * h + 8];
    }
#pragma unroll
    for (int ni = 0; ni < 4; ni++) {
      accN[ni] = __builtin_amdgcn_wmma_f32_16x16x32_bf16(
          false, a.v, false, bn[ni].v, (short)0, accN[ni], false, false);
      accG[ni] = __builtin_amdgcn_wmma_f32_16x16x32_bf16(
          false, a.v, false, bg[ni].v, (short)0, accG[ni], false, false);
    }

    if (more) {
      *(uint4*)&lA[buf ^ 1][row0][col8] = ra0;
      *(uint4*)&lA[buf ^ 1][row1][col8] = ra1;
      *(uint4*)&lB[buf ^ 1][row0][col8] = rb0;
      *(uint4*)&lB[buf ^ 1][row1][col8] = rb1;
    }
    buf ^= 1;
  }

  __bf16* Xo = Xout + (size_t)side * MTOT * Dn;
#pragma unroll
  for (int ni = 0; ni < 4; ni++) {
    int n = n0 + ni * 16 + r;
    float bnv = bias[n], bgv = bias[Dn + n];
#pragma unroll
    for (int v = 0; v < 8; v++) {
      int m = m0 + mWv + 8 * h + v;
      float nl = accN[ni][v] + bnv;
      nl = nl > 0.f ? nl : 0.f;
      float g = 1.f / (1.f + __expf(-(accG[ni][v] + bgv)));
      float xv = bf2f(Xs[(size_t)m * Dn + n]);
      float y = g * xv + (1.f - g) * nl;
      if (final_layer) {
        size_t o = (size_t)m * (2 * Dn) + (size_t)side * Dn + n;
        __builtin_nontemporal_store(y, &out[o]);                       // all_layers
        __builtin_nontemporal_store(y, &out[(size_t)MTOT * 2 * Dn + o]); // out
      } else {
        Xo[(size_t)m * Dn + n] = f2bf(y);
      }
    }
  }
}

// ---------------- launch ----------------
extern "C" void kernel_launch(void* const* d_in, const int* in_sizes, int n_in,
                              void* d_out, int out_size, void* d_ws, size_t ws_size,
                              hipStream_t stream) {
  const float* inputs  = (const float*)d_in[0];
  const float* lp      = (const float*)d_in[1];
  const float* rp      = (const float*)d_in[2];
  const float* left_W  = (const float*)d_in[3];
  const float* left_b  = (const float*)d_in[4];
  const float* right_W = (const float*)d_in[5];
  const float* right_b = (const float*)d_in[6];
  const float* lhwW    = (const float*)d_in[7];
  const float* lhwB    = (const float*)d_in[8];
  const float* rhwW    = (const float*)d_in[9];
  const float* rhwB    = (const float*)d_in[10];
  float* out = (float*)d_out;

  char* ws = (char*)d_ws;
  __bf16* P    = (__bf16*)(ws);                                   size_t o = (size_t)Bn * SP * Dn * 2;
  __bf16* Wlb  = (__bf16*)(ws + o);  o += (size_t)Dn * INSZ * 2;
  __bf16* Wrb  = (__bf16*)(ws + o);  o += (size_t)Dn * INSZ * 2;
  __bf16* HWl  = (__bf16*)(ws + o);  o += (size_t)2 * 2 * Dn * Dn * 2;
  __bf16* HWr  = (__bf16*)(ws + o);  o += (size_t)2 * 2 * Dn * Dn * 2;
  __bf16* X0   = (__bf16*)(ws + o);  o += (size_t)2 * MTOT * Dn * 2;
  __bf16* X1   = (__bf16*)(ws + o);

  // 1) pad + convert activations, convert all weights to bf16
  {
    int n = Bn * SP * Dn;
    pad_kernel<<<(n + 255) / 256, 256, 0, stream>>>(inputs, lp, rp, P);
  }
  { int n = Dn * INSZ;       cvt_f32_bf16<<<(n + 255) / 256, 256, 0, stream>>>(left_W,  Wlb, n); }
  { int n = Dn * INSZ;       cvt_f32_bf16<<<(n + 255) / 256, 256, 0, stream>>>(right_W, Wrb, n); }
  { int n = 2 * 2 * Dn * Dn; cvt_f32_bf16<<<(n + 255) / 256, 256, 0, stream>>>(lhwW, HWl, n); }
  { int n = 2 * 2 * Dn * Dn; cvt_f32_bf16<<<(n + 255) / 256, 256, 0, stream>>>(rhwW, HWr, n); }

  // 2) windowed projection GEMM + ReLU (both sides)
  win_gemm<<<dim3(MTOT / 128, Dn / 128, 2), 256, 0, stream>>>(
      P, Wlb, Wrb, left_b, right_b, X0);

  // 3) highway layers (sequential, ping-pong)
  highway_layer<<<dim3(MTOT / 128, Dn / 64, 2), 256, 0, stream>>>(
      X0, HWl, HWr, lhwB, rhwB, X1, out, 0);
  highway_layer<<<dim3(MTOT / 128, Dn / 64, 2), 256, 0, stream>>>(
      X1, HWl + (size_t)2 * Dn * Dn, HWr + (size_t)2 * Dn * Dn,
      lhwB + 2 * Dn, rhwB + 2 * Dn, X0, out, 1);
}
